// WindowAttention_6854767804423
// MI455X (gfx1250) — matrix operand order repaired and checked by reading, not compiled, not run on hardware
//
#include <hip/hip_runtime.h>
#include <hip/hip_bf16.h>

// ---------------------------------------------------------------------------
// Fused Swin window attention for gfx1250 (MI455X).
//   prep kernel : f32 weights -> f16 (once per launch, into d_ws) and
//                 bias_table[rel_index] -> expanded bias[h][i][j] f32.
//   main kernel : one workgroup (256 thr = 8 wave32) per 49-token window,
//                 fully in LDS: x->f16, QKV GEMM (WMMA f16), per-head
//                 QK^T + bias + mask + softmax + PV (WMMA f16), proj GEMM.
//                 f32 accumulation throughout.
// ---------------------------------------------------------------------------

typedef __attribute__((ext_vector_type(16))) _Float16 v16h;
typedef __attribute__((ext_vector_type(4)))  _Float16 f16x4;
typedef __attribute__((ext_vector_type(8)))  float    v8f;
typedef __attribute__((ext_vector_type(4)))  float    f32x4;

#define NTOK   49
#define NPAD   64
#define CDIM   192
#define NHEADS 6
#define HD     32
#define QKVC   576   // 3 * CDIM

// Workspace layout (bytes):
#define WS_QKVW_OFF  0                         // f16 [576][192] = 221184
#define WS_PROJW_OFF 221184                    // f16 [192][192] =  73728
#define WS_BIAS_OFF  (221184 + 73728)          // f32 [6][49][49] = 57624
#define WS_BYTES     (WS_BIAS_OFF + 6 * 49 * 49 * 4)

// Dynamic LDS layout (bytes):
//   [0,      24576)  : Xh  f16[64][192]   (stage 1)   -- reused as:
//                      Sb  f32[64][64] (16384) + Pb f16[64][64] (8192)
//   [24576,  98304)  : QKV f16[64][576]
//   [98304, 122880)  : Ob  f16[64][192]
#define SMEM_BYTES (24576 + 73728 + 24576)

// ---- WMMA fragment loaders (wave32; contiguous k = 16*h + s in both A and B) -----------

// A fragment (16x32 f16) from row-major f16 [M][>=K], leading dim ld.
// a[0..15] = K = k0 + 16*hf + 0..15
__device__ __forceinline__ v16h load_a_rm(const _Float16* base, int row0, int k0,
                                          int ld, int lane) {
  const int hf = (lane >> 4) & 1;
  const int r  = lane & 15;
  const _Float16* p0 = base + (row0 + r) * ld + k0 + 16 * hf;
  v16h a;
#pragma unroll
  for (int i = 0; i < 16; ++i) a[i] = p0[i];
  return a;
}

// B fragment (32x16 f16) where B^T is row-major f16 [N][>=K] in LDS.
__device__ __forceinline__ v16h load_bT_rm(const _Float16* base, int n0, int k0,
                                           int ld, int lane) {
  const int kh = (lane >> 4) & 1;
  const int c  = lane & 15;
  const _Float16* p = base + (n0 + c) * ld + k0 + 16 * kh;
  v16h b;
#pragma unroll
  for (int i = 0; i < 16; ++i) b[i] = p[i];
  return b;
}

// B fragment (32x16 f16) from row-major f16 [K][N] in LDS, leading dim ld.
__device__ __forceinline__ v16h load_b_rm(const _Float16* base, int k0, int n0,
                                          int ld, int lane) {
  const int kh = (lane >> 4) & 1;
  const int c  = lane & 15;
  v16h b;
#pragma unroll
  for (int p = 0; p < 8; ++p) {
    const int k = k0 + 16 * kh + 2 * p;
    b[2 * p]     = base[(k)     * ld + n0 + c];
    b[2 * p + 1] = base[(k + 1) * ld + n0 + c];
  }
  return b;
}

// B fragment from pre-converted f16 weights Wh[Nout][Kin] in global memory:
// B[k][n] = Wh[n0+n][k]. 16 consecutive halfs per lane -> 2x global_load_b128,
// no conversion in the GEMM inner loop.
__device__ __forceinline__ v16h load_b_wh(const _Float16* __restrict__ Wh, int n0,
                                          int k0, int ldk, int lane) {
  const int kh = (lane >> 4) & 1;
  const int c  = lane & 15;
  const _Float16* p = Wh + (size_t)(n0 + c) * ldk + k0 + 16 * kh;
  v16h b;
#pragma unroll
  for (int i = 0; i < 16; ++i) b[i] = p[i];
  return b;
}

#define WMMA_F16(A, B, C) \
  __builtin_amdgcn_wmma_f32_16x16x32_f16(false, (A), false, (B), (short)0, (C), false, false)

// ---------------------------------------------------------------------------
// Prep: weights -> f16 into ws, bias gather -> expanded f32 table in ws.
// ---------------------------------------------------------------------------
__global__ __launch_bounds__(256)
void swin_prep_kernel(const float* __restrict__ qkv_w,
                      const float* __restrict__ proj_w,
                      const float* __restrict__ bias_table,
                      const int*   __restrict__ rel_index,
                      _Float16* __restrict__ qkv_wh,
                      _Float16* __restrict__ proj_wh,
                      float* __restrict__ biasx) {
  const int tid    = blockIdx.x * blockDim.x + threadIdx.x;
  const int stride = gridDim.x * blockDim.x;
  for (int t = tid; t < QKVC * CDIM; t += stride)
    qkv_wh[t] = (_Float16)qkv_w[t];
  for (int t = tid; t < CDIM * CDIM; t += stride)
    proj_wh[t] = (_Float16)proj_w[t];
  for (int t = tid; t < NHEADS * NTOK * NTOK; t += stride) {
    const int h  = t / (NTOK * NTOK);
    const int ij = t % (NTOK * NTOK);
    biasx[t] = bias_table[rel_index[ij] * NHEADS + h];
  }
}

// ---------------------------------------------------------------------------

__global__ __launch_bounds__(256)
void swin_window_attention_kernel(const float* __restrict__ x,
                                  const float* __restrict__ mask,
                                  const _Float16* __restrict__ qkv_wh,
                                  const float* __restrict__ qkv_b,
                                  const _Float16* __restrict__ proj_wh,
                                  const float* __restrict__ proj_b,
                                  const float* __restrict__ biasx,
                                  float* __restrict__ out) {
  extern __shared__ char smem_raw[];
  _Float16* Xh  = (_Float16*)(smem_raw);                 // [64][192]
  float*    Sb  = (float*)(smem_raw);                    // [64][64] (reuses Xh)
  _Float16* Pb  = (_Float16*)(smem_raw + 64 * 64 * 4);   // [64][64]
  _Float16* QKV = (_Float16*)(smem_raw + 24576);         // [64][576]
  _Float16* Ob  = (_Float16*)(smem_raw + 24576 + 73728); // [64][192]

  const int b    = blockIdx.x;
  const int tid  = threadIdx.x;
  const int wid  = tid >> 5;
  const int lane = tid & 31;
  const int hf   = (lane >> 4) & 1;
  const int lc   = lane & 15;

  // ---- Stage 0: x window -> f16 LDS (b128 loads, packed f16 stores) -------
  const float* xb = x + (size_t)b * NTOK * CDIM;
  for (int i = tid * 4; i < NPAD * CDIM; i += 256 * 4) {
    const int m = i / CDIM;
    f16x4 hv;
    if (m < NTOK) {
      const f32x4 fv = *(const f32x4*)(xb + i);
#pragma unroll
      for (int j = 0; j < 4; ++j) hv[j] = (_Float16)fv[j];
    } else {
#pragma unroll
      for (int j = 0; j < 4; ++j) hv[j] = (_Float16)0.0f;
    }
    *(f16x4*)(Xh + i) = hv;
  }
  __syncthreads();

  // ---- Stage 1: QKV = Xh @ qkv_w^T + qkv_b (q pre-scaled) -----------------
  const float scale = 0.17677669529663687f;  // 32^-0.5
  for (int nt = wid; nt < QKVC / 16; nt += 8) {
    v8f acc[4] = {};
#pragma unroll
    for (int ks = 0; ks < 6; ++ks) {
      const v16h bf = load_b_wh(qkv_wh, nt * 16, ks * 32, CDIM, lane);
#pragma unroll
      for (int mt = 0; mt < 4; ++mt) {
        const v16h af = load_a_rm(Xh, mt * 16, ks * 32, CDIM, lane);
        acc[mt] = WMMA_F16(af, bf, acc[mt]);
      }
    }
    const int col  = nt * 16 + lc;
    const float bb = qkv_b[col];
    const bool isQ = (col < CDIM);
#pragma unroll
    for (int mt = 0; mt < 4; ++mt) {
#pragma unroll
      for (int v = 0; v < 8; ++v) {
        const int row = mt * 16 + v + 8 * hf;
        float val = acc[mt][v] + bb;
        if (isQ) val *= scale;
        if (row >= NTOK) val = 0.0f;  // keep padded K/V rows exactly zero
        QKV[row * QKVC + col] = (_Float16)val;
      }
    }
  }
  __syncthreads();

  // ---- Stage 2: per-head attention ----------------------------------------
  const float* maskw = mask + (size_t)(b & 63) * NTOK * NTOK;
  for (int h = 0; h < NHEADS; ++h) {
    const float* biash = biasx + (size_t)h * NTOK * NTOK;
    // S = Qh @ Kh^T (+ bias + mask): 16 tiles, K = 32 -> one WMMA each.
    for (int t = wid; t < 16; t += 8) {
      const int mt = t & 3, nt = t >> 2;
      const v16h af = load_a_rm (QKV, mt * 16, h * HD, QKVC, lane);
      const v16h bf = load_bT_rm(QKV, nt * 16, CDIM + h * HD, QKVC, lane);
      v8f acc = {};
      acc = WMMA_F16(af, bf, acc);
#pragma unroll
      for (int v = 0; v < 8; ++v) {
        const int i = mt * 16 + v + 8 * hf;
        const int j = nt * 16 + lc;
        float s = acc[v];
        if (i < NTOK && j < NTOK)
          s += biash[i * NTOK + j] + maskw[i * NTOK + j];
        Sb[i * 64 + j] = s;
      }
    }
    __syncthreads();

    // Row softmax (f32), write probs as f16; padded cols/rows -> 0.
    if (tid < NPAD) {
      if (tid < NTOK) {
        float mx = -1e30f;
        for (int j = 0; j < NTOK; ++j) mx = fmaxf(mx, Sb[tid * 64 + j]);
        float sum = 0.0f;
        for (int j = 0; j < NTOK; ++j) sum += __expf(Sb[tid * 64 + j] - mx);
        const float inv = 1.0f / sum;
        for (int j = 0; j < NTOK; ++j)
          Pb[tid * 64 + j] = (_Float16)(__expf(Sb[tid * 64 + j] - mx) * inv);
        for (int j = NTOK; j < 64; ++j) Pb[tid * 64 + j] = (_Float16)0.0f;
      } else {
        for (int j = 0; j < 64; ++j) Pb[tid * 64 + j] = (_Float16)0.0f;
      }
    }
    __syncthreads();

    // O_h = P @ V_h : 8 tiles (4 row x 2 col), K = 64 -> 2 WMMA k-steps.
    {
      const int mt = wid >> 1, nt = wid & 1;
      v8f acc = {};
#pragma unroll
      for (int ks = 0; ks < 2; ++ks) {
        const v16h af = load_a_rm(Pb, mt * 16, ks * 32, 64, lane);
        const v16h bf = load_b_rm(QKV + 2 * CDIM + h * HD + nt * 16,
                                  ks * 32, 0, QKVC, lane);
        acc = WMMA_F16(af, bf, acc);
      }
#pragma unroll
      for (int v = 0; v < 8; ++v) {
        const int row = mt * 16 + v + 8 * hf;
        Ob[row * CDIM + h * HD + nt * 16 + lc] = (_Float16)acc[v];
      }
    }
    __syncthreads();  // Sb/Pb reused by next head
  }

  // ---- Stage 3: out = Ob @ proj_w^T + proj_b ------------------------------
  for (int nt = wid; nt < CDIM / 16; nt += 8) {
    v8f acc[4] = {};
#pragma unroll
    for (int ks = 0; ks < 6; ++ks) {
      const v16h bf = load_b_wh(proj_wh, nt * 16, ks * 32, CDIM, lane);
#pragma unroll
      for (int mt = 0; mt < 4; ++mt) {
        const v16h af = load_a_rm(Ob, mt * 16, ks * 32, CDIM, lane);
        acc[mt] = WMMA_F16(af, bf, acc[mt]);
      }
    }
    const int col  = nt * 16 + lc;
    const float pb = proj_b[col];
    float* ob = out + (size_t)b * NTOK * CDIM;
#pragma unroll
    for (int mt = 0; mt < 4; ++mt) {
#pragma unroll
      for (int v = 0; v < 8; ++v) {
        const int row = mt * 16 + v + 8 * hf;
        if (row < NTOK) ob[row * CDIM + col] = acc[mt][v] + pb;
      }
    }
  }
}

// ---------------------------------------------------------------------------

extern "C" void kernel_launch(void* const* d_in, const int* in_sizes, int n_in,
                              void* d_out, int out_size, void* d_ws, size_t ws_size,
                              hipStream_t stream) {
  const float* x          = (const float*)d_in[0];
  const float* mask       = (const float*)d_in[1];
  const float* qkv_w      = (const float*)d_in[2];
  const float* qkv_b      = (const float*)d_in[3];
  const float* proj_w     = (const float*)d_in[4];
  const float* proj_b     = (const float*)d_in[5];
  const float* bias_table = (const float*)d_in[6];
  const int*   rel_index  = (const int*)d_in[7];
  float* out = (float*)d_out;

  _Float16* qkv_wh  = (_Float16*)((char*)d_ws + WS_QKVW_OFF);
  _Float16* proj_wh = (_Float16*)((char*)d_ws + WS_PROJW_OFF);
  float*    biasx   = (float*)((char*)d_ws + WS_BIAS_OFF);

  // Prep runs every launch (deterministic; ws is scratch, never assumed warm).
  swin_prep_kernel<<<dim3(64), dim3(256), 0, stream>>>(
      qkv_w, proj_w, bias_table, rel_index, qkv_wh, proj_wh, biasx);

  // 120 KB dynamic LDS per workgroup (CDNA5 WGP has 320 KB -> 2 WGs/WGP).
  (void)hipFuncSetAttribute((const void*)swin_window_attention_kernel,
                            hipFuncAttributeMaxDynamicSharedMemorySize,
                            SMEM_BYTES);

  swin_window_attention_kernel<<<dim3(4096), dim3(256), SMEM_BYTES, stream>>>(
      x, mask, qkv_wh, qkv_b, proj_wh, proj_b, biasx, out);
}
